// PointPillarErmvp_14697378087068
// MI455X (gfx1250) — matrix operand order repaired and checked by reading, not compiled, not run on hardware
//
#include <hip/hip_runtime.h>
#include <hip/hip_bf16.h>

typedef __attribute__((ext_vector_type(16))) _Float16 v16h;
typedef __attribute__((ext_vector_type(8)))  float    v8f;
typedef unsigned long long u64;

// ---- problem constants (static shapes from the reference) ----
constexpr int NB = 8;        // N batches (CAVs)
constexpr int CC = 256;      // channels
constexpr int LL = 35200;    // H*W = 100*352
constexpr int LP = 65536;    // L padded to pow2 for bitonic
constexpr int KK = 3520;     // top-k tokens  (ceil(L*0.1))
constexpr int KP = 4096;     // K padded
constexpr int MM = 880;      // clusters      (ceil(K*0.25))
constexpr int KT_I = KK / 16;  // 220 i-tiles
constexpr int MT_I = MM / 16;  // 55 center tiles
constexpr int LT_I = LL / 16;  // 2200 pixel tiles

// ---------------- WMMA helpers ----------------
__device__ inline v8f wmma16(v16h a, v16h b, v8f c) {
  // D = A(16x32 f16) * B(32x16 f16) + C(16x16 f32)
  return __builtin_amdgcn_wmma_f32_16x16x32_f16(false, a, false, b, (short)0, c, false, false);
}
// A-fragment element e -> K offset (contiguous k = half*16 + e, same order as loadB_h, lanes split by `half`)
__device__ inline int kA(int e, int half) { return (half << 4) + e; }

// load A fragment row from an f32 row pointer (row = lane&15 chosen by caller)
__device__ inline v16h loadA_f32(const float* __restrict__ rowp, int k0, int half) {
  v16h a;
#pragma unroll
  for (int e = 0; e < 16; ++e) a[e] = (_Float16)rowp[k0 + kA(e, half)];
  return a;
}
__device__ inline v16h loadA_h(const _Float16* __restrict__ rowp, int k0, int half) {
  v16h a;
#pragma unroll
  for (int e = 0; e < 16; ++e) a[e] = rowp[k0 + kA(e, half)];
  return a;
}
// B fragment: column = lane&15 (caller supplies that row pointer of S), K = half*16+e (contiguous)
__device__ inline v16h loadB_h(const _Float16* __restrict__ rowp, int k0, int half) {
  v16h b;
#pragma unroll
  for (int e = 0; e < 16; ++e) b[e] = rowp[k0 + (half << 4) + e];
  return b;
}

__device__ inline void top10_insert(float (&t)[10], float v) {
  if (v < t[9]) {
    t[9] = v;
#pragma unroll
    for (int i = 9; i > 0; --i) {
      if (t[i] < t[i - 1]) { float tmp = t[i]; t[i] = t[i - 1]; t[i - 1] = tmp; }
    }
  }
}

__device__ inline bool is_ego_n(const int* __restrict__ rl, int B, int n) {
  int o = 0;
  for (int i = 0; i < B; ++i) { if (n == o) return true; o += rl[i]; }
  return false;
}

// ============ 1) fused score MLP -> sortable keys ============
__global__ __launch_bounds__(256) void score_kernel(
    const float* __restrict__ x, const float* __restrict__ dis,
    const float* __restrict__ w1, const float* __restrict__ b1,
    const float* __restrict__ w2, const float* __restrict__ b2,
    u64* __restrict__ sortbuf) {
  const int wave = threadIdx.x >> 5;
  const int lane = threadIdx.x & 31;
  const int half = lane >> 4;
  const int coln = lane & 15;
  const int n  = blockIdx.x / (LL / 128);
  const int lb = blockIdx.x % (LL / 128);
  const int l  = lb * 128 + wave * 16 + coln;
  const float* xn = x + (size_t)n * CC * LL;

  float partial = 0.f;
  for (int ot = 0; ot < 16; ++ot) {
    v8f acc = {};
#pragma unroll
    for (int kt = 0; kt < 8; ++kt) {
      v16h a = loadA_f32(w1 + (size_t)(ot * 16 + coln) * CC, kt * 32, half);
      v16h b;
#pragma unroll
      for (int e = 0; e < 16; ++e)
        b[e] = (_Float16)xn[(size_t)(kt * 32 + (half << 4) + e) * LL + l];
      acc = wmma16(a, b, acc);
    }
#pragma unroll
    for (int r = 0; r < 8; ++r) {
      int o = ot * 16 + r + 8 * half;
      float h = fmaxf(acc[r] + b1[o], 0.f);
      partial += w2[o] * h;
    }
  }
  float tot = partial + __shfl_xor(partial, 16, 32);
  if (half == 0) {
    float s = 1.f / (1.f + __expf(-(tot + b2[0])));
    float wgt = s * dis[(size_t)n * LL + l];   // >= 0, float bits monotone
    sortbuf[(size_t)n * LP + l] =
        (((u64)__float_as_uint(wgt)) << 32) | (u64)(0xFFFFFFFFu - (unsigned)l);
  }
}

__global__ void pad_fill_kernel(u64* __restrict__ sortbuf) {
  size_t t = (size_t)blockIdx.x * blockDim.x + threadIdx.x;
  if (t >= (size_t)NB * LP) return;
  if ((t & (LP - 1)) >= (unsigned)LL) sortbuf[t] = 0ull;
}

// ============ 2) global bitonic sort (descending), 65536 per batch ============
__global__ void bitonic_global(u64* __restrict__ keys, int k, int j) {
  int t = blockIdx.x * blockDim.x + threadIdx.x;   // 0 .. NB*LP-1
  int seg = t >> 16, li = t & (LP - 1);
  int l = li ^ j;
  if (l > li) {
    u64* base = keys + ((size_t)seg << 16);
    u64 a = base[li], b = base[l];
    bool up = ((li & k) == 0);
    if (up ? (a < b) : (a > b)) { base[li] = b; base[l] = a; }
  }
}

// ============ 3) gather topk tokens -> src (f32+f16), sq, conf, idx ============
__global__ __launch_bounds__(256) void gather_kernel(
    const float* __restrict__ x, const u64* __restrict__ sortbuf,
    float* __restrict__ srcf, _Float16* __restrict__ srch,
    float* __restrict__ sq, float* __restrict__ conf, int* __restrict__ sort_idx) {
  const int n = blockIdx.x / KK, k = blockIdx.x % KK;
  const u64 key = sortbuf[(size_t)n * LP + k];
  const unsigned idx = 0xFFFFFFFFu - (unsigned)(key & 0xFFFFFFFFu);
  const float cf = __uint_as_float((unsigned)(key >> 32));
  const int c = threadIdx.x;
  float v = x[((size_t)n * CC + c) * LL + idx] * cf;
  srcf[((size_t)n * KK + k) * CC + c] = v;
  srch[((size_t)n * KK + k) * CC + c] = (_Float16)v;
  __shared__ float red[256];
  red[c] = v * v;
  __syncthreads();
  for (int s = 128; s > 0; s >>= 1) { if (c < s) red[c] += red[c + s]; __syncthreads(); }
  if (c == 0) {
    sq[n * KK + k] = red[0];
    conf[n * KK + k] = cf;
    sort_idx[n * KK + k] = (int)idx;
  }
}

__global__ void init_distmax(unsigned* dm) { if (threadIdx.x == 0 && blockIdx.x == 0) *dm = 0u; }

// ============ 4) knn density (top-10 smallest dist per row) + global dist max ============
__global__ __launch_bounds__(32) void knn_kernel(
    const _Float16* __restrict__ srch, const float* __restrict__ sq,
    float* __restrict__ density, unsigned* __restrict__ distmax) {
  const int n = blockIdx.x / KT_I, it = blockIdx.x % KT_I;
  const int lane = threadIdx.x & 31, half = lane >> 4, col = lane & 15;
  const int i = it * 16 + col;
  const _Float16* Sn = srch + (size_t)n * KK * CC;
  const float sqi = sq[n * KK + i];
  const _Float16* Brow = Sn + (size_t)i * CC;

  float t[10];
#pragma unroll
  for (int q = 0; q < 10; ++q) t[q] = 3.0e38f;
  float vmax = 0.f;

  for (int jt = 0; jt < KT_I; ++jt) {
    const _Float16* Arow = Sn + (size_t)(jt * 16 + col) * CC;
    if (jt + 1 < KT_I) __builtin_prefetch(Sn + (size_t)((jt + 1) * 16 + col) * CC, 0, 0);
    v8f acc = {};
#pragma unroll
    for (int kt = 0; kt < 8; ++kt)
      acc = wmma16(loadA_h(Arow, kt * 32, half), loadB_h(Brow, kt * 32, half), acc);
#pragma unroll
    for (int r = 0; r < 8; ++r) {
      int j = jt * 16 + r + 8 * half;
      float d = fmaxf(sqi + sq[n * KK + j] - 2.0f * acc[r], 0.f);
      vmax = fmaxf(vmax, d);
      top10_insert(t, d);
    }
  }
  float orig[10];
#pragma unroll
  for (int q = 0; q < 10; ++q) orig[q] = t[q];
#pragma unroll
  for (int q = 0; q < 10; ++q) top10_insert(t, __shfl_xor(orig[q], 16, 32));

  if (half == 0) {
    float mean = 0.f;
#pragma unroll
    for (int q = 0; q < 10; ++q) mean += t[q];
    density[n * KK + i] = __expf(-mean * 0.1f) + (float)i * 1e-6f;
  }
  atomicMax(distmax, __float_as_uint(vmax));
}

// ============ 5) delta = min dist to higher-density token; score = delta*density ============
__global__ __launch_bounds__(32) void delta_kernel(
    const _Float16* __restrict__ srch, const float* __restrict__ sq,
    const float* __restrict__ density, const unsigned* __restrict__ distmax,
    float* __restrict__ scorearr) {
  const int n = blockIdx.x / KT_I, it = blockIdx.x % KT_I;
  const int lane = threadIdx.x & 31, half = lane >> 4, col = lane & 15;
  const int i = it * 16 + col;
  const _Float16* Sn = srch + (size_t)n * KK * CC;
  const float sqi = sq[n * KK + i];
  const float di = density[n * KK + i];
  const float dmax = __uint_as_float(*distmax);
  const _Float16* Brow = Sn + (size_t)i * CC;
  float best = dmax;

  for (int jt = 0; jt < KT_I; ++jt) {
    const _Float16* Arow = Sn + (size_t)(jt * 16 + col) * CC;
    if (jt + 1 < KT_I) __builtin_prefetch(Sn + (size_t)((jt + 1) * 16 + col) * CC, 0, 0);
    v8f acc = {};
#pragma unroll
    for (int kt = 0; kt < 8; ++kt)
      acc = wmma16(loadA_h(Arow, kt * 32, half), loadB_h(Brow, kt * 32, half), acc);
#pragma unroll
    for (int r = 0; r < 8; ++r) {
      int j = jt * 16 + r + 8 * half;
      float d = fmaxf(sqi + sq[n * KK + j] - 2.0f * acc[r], 0.f);
      if (density[n * KK + j] > di) best = fminf(best, d);
    }
  }
  best = fminf(best, __shfl_xor(best, 16, 32));
  if (half == 0) scorearr[n * KK + i] = best * di;
}

// ============ 6) build score keys + in-LDS bitonic for M=880 centers ============
__global__ void scorekeys_kernel(const float* __restrict__ scorearr, u64* __restrict__ scorebuf) {
  int t = blockIdx.x * blockDim.x + threadIdx.x;
  if (t >= NB * KP) return;
  int n = t / KP, k = t % KP;
  u64 key = 0ull;
  if (k < KK)
    key = (((u64)__float_as_uint(scorearr[n * KK + k])) << 32) |
          (u64)(0xFFFFFFFFu - (unsigned)k);
  scorebuf[t] = key;
}

__global__ __launch_bounds__(1024) void center_sort_kernel(
    const u64* __restrict__ scorebuf, int* __restrict__ center_idx) {
  __shared__ u64 sk[KP];
  const int n = blockIdx.x;
  for (int i = threadIdx.x; i < KP; i += blockDim.x) sk[i] = scorebuf[(size_t)n * KP + i];
  __syncthreads();
  for (int k = 2; k <= KP; k <<= 1)
    for (int j = k >> 1; j > 0; j >>= 1) {
      for (int i = threadIdx.x; i < KP; i += blockDim.x) {
        int l = i ^ j;
        if (l > i) {
          u64 a = sk[i], b = sk[l];
          bool up = ((i & k) == 0);
          if (up ? (a < b) : (a > b)) { sk[i] = b; sk[l] = a; }
        }
      }
      __syncthreads();
    }
  for (int m = threadIdx.x; m < MM; m += blockDim.x)
    center_idx[n * MM + m] = (int)(0xFFFFFFFFu - (unsigned)sk[m]);
}

// ============ 7) assign each token to nearest center (argmin, low index wins) ============
__global__ __launch_bounds__(32) void assign_kernel(
    const _Float16* __restrict__ srch, const float* __restrict__ sq,
    const int* __restrict__ center_idx, int* __restrict__ idx_cluster) {
  const int n = blockIdx.x / KT_I, it = blockIdx.x % KT_I;
  const int lane = threadIdx.x & 31, half = lane >> 4, col = lane & 15;
  const int i = it * 16 + col;
  const _Float16* Sn = srch + (size_t)n * KK * CC;
  const float sqi = sq[n * KK + i];
  const _Float16* Brow = Sn + (size_t)i * CC;
  float bestv = 3.0e38f;
  int bestm = 0;

  for (int mt = 0; mt < MT_I; ++mt) {
    const int arow_tok = center_idx[n * MM + mt * 16 + col];
    const _Float16* Arow = Sn + (size_t)arow_tok * CC;
    v8f acc = {};
#pragma unroll
    for (int kt = 0; kt < 8; ++kt)
      acc = wmma16(loadA_h(Arow, kt * 32, half), loadB_h(Brow, kt * 32, half), acc);
#pragma unroll
    for (int r = 0; r < 8; ++r) {
      int m = mt * 16 + r + 8 * half;
      int ctok = center_idx[n * MM + m];
      float d = fmaxf(sqi + sq[n * KK + ctok] - 2.0f * acc[r], 0.f);
      if (d < bestv || (d == bestv && m < bestm)) { bestv = d; bestm = m; }
    }
  }
  float ov = __shfl_xor(bestv, 16, 32);
  int   om = __shfl_xor(bestm, 16, 32);
  if (ov < bestv || (ov == bestv && om < bestm)) { bestv = ov; bestm = om; }
  if (half == 0) idx_cluster[n * KK + i] = bestm;
}

__global__ void center_override_kernel(const int* __restrict__ center_idx,
                                       int* __restrict__ idx_cluster) {
  int t = blockIdx.x * blockDim.x + threadIdx.x;
  if (t >= NB * MM) return;
  int n = t / MM, m = t % MM;
  idx_cluster[n * KK + center_idx[n * MM + m]] = m;
}

// ============ 8) merge tokens (segment sums) ============
__global__ void init_merge_kernel(float* __restrict__ merged, float* __restrict__ all_w) {
  size_t t = (size_t)blockIdx.x * blockDim.x + threadIdx.x;
  if (t >= (size_t)NB * MM * CC) return;
  merged[t] = 0.f;
  if ((t & (CC - 1)) == 0) all_w[t >> 8] = 1e-6f;
}

__global__ __launch_bounds__(256) void accum_kernel(
    const float* __restrict__ srcf, const float* __restrict__ conf,
    const int* __restrict__ idx_cluster, float* __restrict__ merged,
    float* __restrict__ all_w) {
  const int n = blockIdx.x / KK, k = blockIdx.x % KK;
  const int m = idx_cluster[n * KK + k];
  const float cf = conf[n * KK + k];
  const int c = threadIdx.x;
  atomicAdd(&merged[((size_t)n * MM + m) * CC + c],
            srcf[((size_t)n * KK + k) * CC + c] * cf);
  if (c == 0) atomicAdd(&all_w[n * MM + m], cf);
}

__global__ void divide_kernel(float* __restrict__ merged, const float* __restrict__ all_w) {
  size_t t = (size_t)blockIdx.x * blockDim.x + threadIdx.x;
  if (t >= (size_t)NB * MM * CC) return;
  merged[t] /= all_w[t >> 8];
}

// ============ 9) BEV output: ego keeps dense features, others get scattered clusters ============
__global__ void bev_base_kernel(const float* __restrict__ x, const int* __restrict__ rl,
                                int B, float* __restrict__ out) {
  size_t t = (size_t)blockIdx.x * blockDim.x + threadIdx.x;
  if (t >= (size_t)NB * CC * LL) return;
  int n = (int)(t / ((size_t)CC * LL));
  out[t] = is_ego_n(rl, B, n) ? x[t] : 0.f;
}

__global__ __launch_bounds__(256) void scatter_kernel(
    const float* __restrict__ merged, const int* __restrict__ idx_cluster,
    const int* __restrict__ sort_idx, const int* __restrict__ rl, int B,
    float* __restrict__ out) {
  const int n = blockIdx.x / KK, k = blockIdx.x % KK;
  if (is_ego_n(rl, B, n)) return;
  const int m = idx_cluster[n * KK + k];
  const int l = sort_idx[n * KK + k];
  const int c = threadIdx.x;
  out[((size_t)n * CC + c) * LL + l] = merged[((size_t)n * MM + m) * CC + c];
}

// ============ 10) ego heads: stacked [cls(2); reg(14)] = 16x256 A -> one WMMA M-tile ============
__global__ __launch_bounds__(32) void heads_kernel(
    const float* __restrict__ x, const float* __restrict__ cw, const float* __restrict__ cb,
    const float* __restrict__ rw, const float* __restrict__ rb,
    const int* __restrict__ rl, int B, float* __restrict__ psm, float* __restrict__ rm) {
  const int b = blockIdx.x / LT_I, lt = blockIdx.x % LT_I;
  int n = 0;
  for (int i = 0; i < b; ++i) n += rl[i];
  const int lane = threadIdx.x & 31, half = lane >> 4, coln = lane & 15;
  const int l = lt * 16 + coln;
  const float* xn = x + (size_t)n * CC * LL;

  v8f acc = {};
#pragma unroll
  for (int kt = 0; kt < 8; ++kt) {
    v16h a, bf;
#pragma unroll
    for (int e = 0; e < 16; ++e) {
      int kk = kt * 32 + kA(e, half);
      a[e] = (_Float16)(coln < 2 ? cw[coln * CC + kk] : rw[(coln - 2) * CC + kk]);
    }
#pragma unroll
    for (int e = 0; e < 16; ++e)
      bf[e] = (_Float16)xn[(size_t)(kt * 32 + (half << 4) + e) * LL + l];
    acc = wmma16(a, bf, acc);
  }
#pragma unroll
  for (int r = 0; r < 8; ++r) {
    int o = r + 8 * half;
    float v = acc[r];
    if (o < 2) psm[((size_t)b * 2 + o) * LL + l] = v + cb[o];
    else       rm[((size_t)b * 14 + (o - 2)) * LL + l] = v + rb[o - 2];
  }
}

// ===================================================================
extern "C" void kernel_launch(void* const* d_in, const int* in_sizes, int n_in,
                              void* d_out, int out_size, void* d_ws, size_t ws_size,
                              hipStream_t stream) {
  const float* x   = (const float*)d_in[0];
  const float* dis = (const float*)d_in[1];
  const float* w1  = (const float*)d_in[2];
  const float* b1  = (const float*)d_in[3];
  const float* w2  = (const float*)d_in[4];
  const float* b2  = (const float*)d_in[5];
  const float* cw  = (const float*)d_in[6];
  const float* cb  = (const float*)d_in[7];
  const float* rw  = (const float*)d_in[8];
  const float* rb  = (const float*)d_in[9];
  const int*   rl  = (const int*)d_in[10];
  const int B = in_sizes[10];                 // 4

  // ---- workspace carve-up ----
  char* wp = (char*)d_ws;
  auto alloc = [&](size_t bytes) { char* p = wp; wp += (bytes + 255) & ~(size_t)255; return p; };
  u64*      sortbuf    = (u64*)alloc((size_t)NB * LP * sizeof(u64));
  u64*      scorebuf   = (u64*)alloc((size_t)NB * KP * sizeof(u64));
  int*      sort_idx   = (int*)alloc((size_t)NB * KK * sizeof(int));
  float*    conf       = (float*)alloc((size_t)NB * KK * sizeof(float));
  float*    sq         = (float*)alloc((size_t)NB * KK * sizeof(float));
  float*    density    = (float*)alloc((size_t)NB * KK * sizeof(float));
  float*    scorearr   = (float*)alloc((size_t)NB * KK * sizeof(float));
  unsigned* distmax    = (unsigned*)alloc(256);
  int*      center_idx = (int*)alloc((size_t)NB * MM * sizeof(int));
  int*      idx_clu    = (int*)alloc((size_t)NB * KK * sizeof(int));
  float*    all_w      = (float*)alloc((size_t)NB * MM * sizeof(float));
  float*    merged     = (float*)alloc((size_t)NB * MM * CC * sizeof(float));
  float*    srcf       = (float*)alloc((size_t)NB * KK * CC * sizeof(float));
  _Float16* srch       = (_Float16*)alloc((size_t)NB * KK * CC * sizeof(_Float16));

  float* bev = (float*)d_out;
  float* psm = bev + (size_t)NB * CC * LL;
  float* rm  = psm + (size_t)B * 2 * LL;

  // 1) fused score MLP -> sort keys
  score_kernel<<<NB * (LL / 128), 256, 0, stream>>>(x, dis, w1, b1, w2, b2, sortbuf);
  pad_fill_kernel<<<(NB * LP + 255) / 256, 256, 0, stream>>>(sortbuf);

  // 2) bitonic sort (descending) per batch
  for (int k = 2; k <= LP; k <<= 1)
    for (int j = k >> 1; j > 0; j >>= 1)
      bitonic_global<<<(NB * LP) / 256, 256, 0, stream>>>(sortbuf, k, j);

  // 3) gather tokens
  gather_kernel<<<NB * KK, 256, 0, stream>>>(x, sortbuf, srcf, srch, sq, conf, sort_idx);

  // 4) knn density + global dist max
  init_distmax<<<1, 32, 0, stream>>>(distmax);
  knn_kernel<<<NB * KT_I, 32, 0, stream>>>(srch, sq, density, distmax);

  // 5) delta & score
  delta_kernel<<<NB * KT_I, 32, 0, stream>>>(srch, sq, density, distmax, scorearr);

  // 6) cluster centers (top-M by score)
  scorekeys_kernel<<<(NB * KP + 255) / 256, 256, 0, stream>>>(scorearr, scorebuf);
  center_sort_kernel<<<NB, 1024, 0, stream>>>(scorebuf, center_idx);

  // 7) nearest-center assignment
  assign_kernel<<<NB * KT_I, 32, 0, stream>>>(srch, sq, center_idx, idx_clu);
  center_override_kernel<<<(NB * MM + 255) / 256, 256, 0, stream>>>(center_idx, idx_clu);

  // 8) weighted cluster merge
  init_merge_kernel<<<(NB * MM * CC + 255) / 256, 256, 0, stream>>>(merged, all_w);
  accum_kernel<<<NB * KK, 256, 0, stream>>>(srcf, conf, idx_clu, merged, all_w);
  divide_kernel<<<(NB * MM * CC + 255) / 256, 256, 0, stream>>>(merged, all_w);

  // 9) BEV scatter + ego dense copy
  bev_base_kernel<<<(int)(((size_t)NB * CC * LL + 255) / 256), 256, 0, stream>>>(x, rl, B, bev);
  scatter_kernel<<<NB * KK, 256, 0, stream>>>(merged, idx_clu, sort_idx, rl, B, bev);

  // 10) ego detection heads (WMMA, stacked 16-row weight matrix)
  heads_kernel<<<B * LT_I, 32, 0, stream>>>(x, cw, cb, rw, rb, rl, B, psm, rm);
}
